// HungarianMatcher_26018911879174
// MI455X (gfx1250) — compile-verified, matrix operand order checked by reading, not run on hardware
//
#include <hip/hip_runtime.h>
#include <math.h>

#define B_   8
#define Q_   300
#define M_   100
#define C1_  81
#define HW_  16384
#define QT   19        // ceil(300/16)
#define MT   7         // ceil(100/16)
#define KSPLIT 16
#define KCH  (HW_ / KSPLIT)   // 1024

#define COST_CLASS_ 1.0f
#define COST_BBOX_  5.0f
#define COST_MASK_  2.0f
#define EPSF        1e-6f

typedef __attribute__((ext_vector_type(16))) _Float16 v16h;
typedef __attribute__((ext_vector_type(8)))  float    v8f;

// fast sigmoid: exp + rcp (v_exp_f32 + v_rcp_f32) instead of IEEE divide chain;
// result is rounded to f16 for the matrix pipe anyway.
__device__ __forceinline__ float sigf(float x) {
    return __builtin_amdgcn_rcpf(1.0f + __expf(-x));
}
__device__ __forceinline__ _Float16 sigh(float x) {
    return (_Float16)sigf(x);
}

// ---------------------------------------------------------------------------
// Row sums: dst[b*rows + r] = sum over HW of (sigmoid(src) or src)
// grid (rows, B), block 256
// ---------------------------------------------------------------------------
__global__ __launch_bounds__(256) void rowsum_kernel(
    const float* __restrict__ src, float* __restrict__ dst,
    int rows, int apply_sigmoid)
{
    const int r = blockIdx.x, b = blockIdx.y, t = threadIdx.x;
    const float* p = src + ((size_t)b * rows + r) * HW_;
    float s = 0.f;
    for (int i = t; i < HW_; i += 256) {
        float x = p[i];
        s += apply_sigmoid ? sigf(x) : x;
    }
    __shared__ float red[256];
    red[t] = s; __syncthreads();
    for (int st = 128; st > 0; st >>= 1) {
        if (t < st) red[t] += red[t + st];
        __syncthreads();
    }
    if (t == 0) dst[(size_t)b * rows + r] = red[0];
}

// ---------------------------------------------------------------------------
// Cheap cost terms: rest[b,q,m] = class + 5*(bbox_l1 - giou)
// grid (Q, B), block 128
// ---------------------------------------------------------------------------
__global__ __launch_bounds__(128) void rest_kernel(
    const float* __restrict__ logits, const float* __restrict__ pboxes,
    const int* __restrict__ glabels,  const float* __restrict__ gboxes,
    float* __restrict__ rest)
{
    const int q = blockIdx.x, b = blockIdx.y, t = threadIdx.x;
    __shared__ float l[C1_];
    __shared__ float red[128];

    const float* lg = logits + ((size_t)b * Q_ + q) * C1_;
    if (t < C1_) l[t] = lg[t];
    __syncthreads();

    red[t] = (t < C1_) ? l[t] : -INFINITY;
    __syncthreads();
    for (int s = 64; s > 0; s >>= 1) {
        if (t < s) red[t] = fmaxf(red[t], red[t + s]);
        __syncthreads();
    }
    const float mx = red[0];
    __syncthreads();

    red[t] = (t < C1_) ? __expf(l[t] - mx) : 0.f;
    __syncthreads();
    for (int s = 64; s > 0; s >>= 1) {
        if (t < s) red[t] += red[t + s];
        __syncthreads();
    }
    const float denom = red[0];

    const float* pb = pboxes + ((size_t)b * Q_ + q) * 4;
    const float p0 = pb[0], p1 = pb[1], p2 = pb[2], p3 = pb[3];

    if (t < M_) {
        const int m = t;
        const int lab = glabels[b * M_ + m];
        const float cc = -(__expf(l[lab] - mx) / denom);

        const float* g = gboxes + ((size_t)b * M_ + m) * 4;
        const float g0 = g[0], g1 = g[1], g2 = g[2], g3 = g[3];

        const float cb = fabsf(p0 - g0) + fabsf(p1 - g1) +
                         fabsf(p2 - g2) + fabsf(p3 - g3);

        const float ltx = fmaxf(p0, g0), lty = fmaxf(p1, g1);
        const float rbx = fminf(p2, g2), rby = fminf(p3, g3);
        const float iw = fmaxf(rbx - ltx, 0.f), ih = fmaxf(rby - lty, 0.f);
        const float inter = iw * ih;
        const float a1 = (p2 - p0) * (p3 - p1);
        const float a2 = (g2 - g0) * (g3 - g1);
        const float uni = a1 + a2 - inter;
        const float iou = inter / (uni + EPSF);
        const float lmx = fminf(p0, g0), lmy = fminf(p1, g1);
        const float rmx = fmaxf(p2, g2), rmy = fmaxf(p3, g3);
        const float ew = fmaxf(rmx - lmx, 0.f), eh = fmaxf(rmy - lmy, 0.f);
        const float am = ew * eh;
        const float giou = iou - (am - uni) / (am + EPSF);

        rest[((size_t)b * Q_ + q) * M_ + m] =
            COST_CLASS_ * cc + COST_BBOX_ * (cb - giou);
    }
}

// ---------------------------------------------------------------------------
// WMMA partial GEMM: part[ks][b][q][m] = sum over K-slice of sig(pm)*gm
// One wave per (qtile, b, kslice). 7 M-tile accumulators held in registers.
// A layout (16-bit 16x32): lane L -> row q0+(L%16); h=L/16 selects K-halves.
// grid (QT, B, KSPLIT), block 32
// ---------------------------------------------------------------------------
__global__ __launch_bounds__(32) void wmma_part_kernel(
    const float* __restrict__ pmasks, const float* __restrict__ gmasks,
    float* __restrict__ part)
{
    const int qt = blockIdx.x, b = blockIdx.y, ks = blockIdx.z;
    const int lane = threadIdx.x;
    const int n = lane & 15;
    const int h = lane >> 4;
    const int q0 = qt * 16;

    int qrow = q0 + n; if (qrow >= Q_) qrow = Q_ - 1;   // clamp pad rows
    const float* pA = pmasks + ((size_t)b * Q_ + qrow) * HW_ + 8 * h;

    const float* pB[MT];
#pragma unroll
    for (int mt = 0; mt < MT; mt++) {
        int mrow = mt * 16 + n; if (mrow >= M_) mrow = M_ - 1;
        pB[mt] = gmasks + ((size_t)b * M_ + mrow) * HW_ + 8 * h;
    }

    const v8f zero = {0.f, 0.f, 0.f, 0.f, 0.f, 0.f, 0.f, 0.f};
    v8f acc[MT];
#pragma unroll
    for (int mt = 0; mt < MT; mt++) acc[mt] = zero;

    const int k0 = ks * KCH, k1 = k0 + KCH;
    for (int kc = k0; kc < k1; kc += 32) {
        const float4 a0 = *(const float4*)(pA + kc);
        const float4 a1 = *(const float4*)(pA + kc + 4);
        const float4 a2 = *(const float4*)(pA + kc + 16);
        const float4 a3 = *(const float4*)(pA + kc + 20);
        v16h av;
        av[0]  = sigh(a0.x); av[1]  = sigh(a0.y); av[2]  = sigh(a0.z); av[3]  = sigh(a0.w);
        av[4]  = sigh(a1.x); av[5]  = sigh(a1.y); av[6]  = sigh(a1.z); av[7]  = sigh(a1.w);
        av[8]  = sigh(a2.x); av[9]  = sigh(a2.y); av[10] = sigh(a2.z); av[11] = sigh(a2.w);
        av[12] = sigh(a3.x); av[13] = sigh(a3.y); av[14] = sigh(a3.z); av[15] = sigh(a3.w);

#pragma unroll
        for (int mt = 0; mt < MT; mt++) {
            const float4 b0 = *(const float4*)(pB[mt] + kc);
            const float4 b1 = *(const float4*)(pB[mt] + kc + 4);
            const float4 b2 = *(const float4*)(pB[mt] + kc + 16);
            const float4 b3 = *(const float4*)(pB[mt] + kc + 20);
            v16h bv;
            bv[0]  = (_Float16)b0.x; bv[1]  = (_Float16)b0.y; bv[2]  = (_Float16)b0.z; bv[3]  = (_Float16)b0.w;
            bv[4]  = (_Float16)b1.x; bv[5]  = (_Float16)b1.y; bv[6]  = (_Float16)b1.z; bv[7]  = (_Float16)b1.w;
            bv[8]  = (_Float16)b2.x; bv[9]  = (_Float16)b2.y; bv[10] = (_Float16)b2.z; bv[11] = (_Float16)b2.w;
            bv[12] = (_Float16)b3.x; bv[13] = (_Float16)b3.y; bv[14] = (_Float16)b3.z; bv[15] = (_Float16)b3.w;

            acc[mt] = __builtin_amdgcn_wmma_f32_16x16x32_f16(
                false, av, false, bv, (short)0, acc[mt], false, false);
        }
    }

    // D layout: lane -> col m0+n ; VGPR r -> row q0 + r + 8*h
    float* pout = part + (size_t)ks * B_ * Q_ * M_;
#pragma unroll
    for (int mt = 0; mt < MT; mt++) {
        const int m = mt * 16 + n;
        if (m >= M_) continue;
#pragma unroll
        for (int r = 0; r < 8; r++) {
            const int q = q0 + r + 8 * h;
            if (q >= Q_) continue;
            pout[((size_t)b * Q_ + q) * M_ + m] = acc[mt][r];
        }
    }
}

// ---------------------------------------------------------------------------
// Final cost: C = rest + 2*(1 - 2*num/(pm_sum + gm_sum + eps))
// Deterministic fixed-order reduction over K-slices.
// ---------------------------------------------------------------------------
__global__ __launch_bounds__(256) void assemble_kernel(
    const float* __restrict__ part, const float* __restrict__ rest,
    const float* __restrict__ pm_sum, const float* __restrict__ gm_sum,
    float* __restrict__ C)
{
    const int idx = blockIdx.x * 256 + threadIdx.x;
    if (idx >= B_ * Q_ * M_) return;
    const int m = idx % M_;
    const int q = (idx / M_) % Q_;
    const int b = idx / (M_ * Q_);
    float num = 0.f;
#pragma unroll
    for (int s = 0; s < KSPLIT; s++)
        num += part[(size_t)s * B_ * Q_ * M_ + idx];
    const float den = pm_sum[b * Q_ + q] + gm_sum[b * M_ + m] + EPSF;
    const float cmask = 1.0f - 2.0f * num / den;
    C[idx] = rest[idx] + COST_MASK_ * cmask;
}

// ---------------------------------------------------------------------------
// Jonker-Volgenant Hungarian on C[b].T (rows = gt M=100, cols = queries Q=300).
// One block per batch; parallel column updates + block argmin reductions.
// double potentials to mirror the float64 reference solver.
// ---------------------------------------------------------------------------
__global__ __launch_bounds__(256) void hungarian_kernel(
    const float* __restrict__ C, int* __restrict__ pidx, int* __restrict__ gidx)
{
    const int b = blockIdx.x;
    const float* Cb = C + (size_t)b * Q_ * M_;
    const int t = threadIdx.x;

    __shared__ double u[M_];
    __shared__ double v[Q_];
    __shared__ double minv[Q_];
    __shared__ int    way[Q_];
    __shared__ int    p[Q_];
    __shared__ int    used[Q_];
    __shared__ double redv[256];
    __shared__ int    redi[256];
    __shared__ int    s_cur, s_j0, s_j1, s_done;
    __shared__ double s_delta;

    for (int j = t; j < Q_; j += 256) { v[j] = 0.0; p[j] = -1; }
    for (int j = t; j < M_; j += 256) u[j] = 0.0;
    __syncthreads();

    for (int i = 0; i < M_; i++) {
        for (int j = t; j < Q_; j += 256) { minv[j] = INFINITY; way[j] = -1; used[j] = 0; }
        if (t == 0) { s_cur = i; s_j0 = -1; s_done = 0; }
        __syncthreads();

        for (;;) {
            const int cr = s_cur, j0 = s_j0;
            const double ucr = u[cr];
            for (int j = t; j < Q_; j += 256) {
                if (!used[j]) {
                    const double cur = (double)Cb[(size_t)j * M_ + cr] - ucr - v[j];
                    if (cur < minv[j]) { minv[j] = cur; way[j] = j0; }
                }
            }
            __syncthreads();

            double bv = INFINITY; int bi = Q_;
            for (int j = t; j < Q_; j += 256) {
                const double val = used[j] ? INFINITY : minv[j];
                if (val < bv) { bv = val; bi = j; }
            }
            redv[t] = bv; redi[t] = bi;
            __syncthreads();
            for (int s = 128; s > 0; s >>= 1) {
                if (t < s) {
                    const double ov = redv[t + s]; const int oi = redi[t + s];
                    if (ov < redv[t] || (ov == redv[t] && oi < redi[t])) {
                        redv[t] = ov; redi[t] = oi;
                    }
                }
                __syncthreads();
            }
            if (t == 0) { s_j1 = redi[0]; s_delta = redv[0]; u[i] += redv[0]; }
            __syncthreads();

            const int j1 = s_j1; const double delta = s_delta;
            for (int j = t; j < Q_; j += 256) {
                if (used[j]) { u[p[j]] += delta; v[j] -= delta; }
                else          minv[j] -= delta;
            }
            __syncthreads();

            if (t == 0) {
                used[j1] = 1; s_j0 = j1;
                const int nr = p[j1];
                s_cur = nr;
                if (nr < 0) s_done = 1;
            }
            __syncthreads();
            if (s_done) break;
        }

        if (t == 0) {
            int j = s_j0;
            while (j != -1) {
                const int jp = way[j];
                p[j] = (jp != -1) ? p[jp] : i;
                j = jp;
            }
        }
        __syncthreads();
    }

    if (t == 0) {
        int cnt = 0;
        for (int q = 0; q < Q_; q++) {
            if (p[q] >= 0) {
                pidx[b * M_ + cnt] = q;
                gidx[b * M_ + cnt] = p[q];
                cnt++;
            }
        }
    }
}

// ---------------------------------------------------------------------------
extern "C" void kernel_launch(void* const* d_in, const int* in_sizes, int n_in,
                              void* d_out, int out_size, void* d_ws, size_t ws_size,
                              hipStream_t stream)
{
    const float* pred_logits = (const float*)d_in[0];
    const float* pred_boxes  = (const float*)d_in[1];
    const float* pred_masks  = (const float*)d_in[2];
    const int*   gt_labels   = (const int*)d_in[3];
    const float* gt_boxes    = (const float*)d_in[4];
    const float* gt_masks    = (const float*)d_in[5];
    // d_in[6] = valid (all ones; unused)

    float* C    = (float*)d_out;                  // [B,Q,M]
    int*   pidx = (int*)d_out + B_ * Q_ * M_;     // [B,M]
    int*   gidx = pidx + B_ * M_;                 // [B,M]

    char* ws = (char*)d_ws;
    float* pm_sum = (float*)ws; ws += sizeof(float) * B_ * Q_;
    float* gm_sum = (float*)ws; ws += sizeof(float) * B_ * M_;
    float* rest   = (float*)ws; ws += sizeof(float) * B_ * Q_ * M_;
    float* part   = (float*)ws; // KSPLIT * B*Q*M floats (~15.4 MB)

    rowsum_kernel<<<dim3(Q_, B_), 256, 0, stream>>>(pred_masks, pm_sum, Q_, 1);
    rowsum_kernel<<<dim3(M_, B_), 256, 0, stream>>>(gt_masks,  gm_sum, M_, 0);
    rest_kernel<<<dim3(Q_, B_), 128, 0, stream>>>(pred_logits, pred_boxes,
                                                  gt_labels, gt_boxes, rest);
    wmma_part_kernel<<<dim3(QT, B_, KSPLIT), 32, 0, stream>>>(pred_masks, gt_masks, part);
    assemble_kernel<<<(B_ * Q_ * M_ + 255) / 256, 256, 0, stream>>>(
        part, rest, pm_sum, gm_sum, C);
    hungarian_kernel<<<B_, 256, 0, stream>>>(C, pidx, gidx);
}
